// ConcatenatedIrrepsTensorProduct_23424751632779
// MI455X (gfx1250) — matrix operand order repaired and checked by reading, not compiled, not run on hardware
//
#include <hip/hip_runtime.h>
#include <hip/hip_bf16.h>

typedef __attribute__((ext_vector_type(16))) _Float16 v16h;
typedef __attribute__((ext_vector_type(8)))  _Float16 v8h;
typedef __attribute__((ext_vector_type(8)))  float    v8f;

// LDS offsets (in halfs) for transposed f16 weights WT[n][k] (scales folded in)
#define O_M1 0
#define O_M2 4096
#define O_M3 8192
#define O_W0 16384
#define O_W1 18432
#define O_W2 20480
#define O_W3 22528
#define O_L0 24576
#define O_L1 26624
#define O_F0 28672
#define O_F1 29696
#define O_SCR 30720          // per-wave D->B bounce scratch: 4 waves x 2 slots x 640 halfs
#define SCR_STRIDE 40        // halfs per column (80B: 16B-aligned, bank-conflict-free)
#define LDS_HALFS (30720 + 4 * 2 * 640)

#define SILU_CST 1.6768f
#define INV_S3   0.57735027f
#define RSQ32    0.17677670f

__device__ __forceinline__ v8f wmma_f16(v16h a, v16h b, v8f c) {
  return __builtin_amdgcn_wmma_f32_16x16x32_f16(false, a, false, b, (short)0, c,
                                                false, false);
}

// A-fragment (16x32 f16) of a transposed weight matrix stored row-major in LDS.
__device__ __forceinline__ v16h lds_afrag(const _Float16* base, int rowlen,
                                          int mt, int kc, int n, int hi) {
  const _Float16* p = base + (mt * 16 + n) * rowlen + kc * 32 + hi * 16;
  v8h lo = *(const v8h*)(p);
  v8h hh = *(const v8h*)(p + 8);
  return __builtin_shufflevector(lo, hh, 0, 1, 2, 3, 4, 5, 6, 7, 8, 9, 10, 11,
                                 12, 13, 14, 15);
}

// Convert a pair of 16x16 f32 D tiles (features 0..31 of a K-chunk, columns =
// events) into the B fragment (32x16 f16) of the next GEMM via a tiny LDS
// bounce. Lane (n,hi) owns rows hi*8..hi*8+7 of each tile; stores them at
// column n, then reloads the contiguous K = hi*16..hi*16+15 run it needs.
__device__ __forceinline__ v16h chunkToB(v8f t0, v8f t1, _Float16* sb, int n,
                                         int hi) {
  v8h h0, h1;
#pragma unroll
  for (int j = 0; j < 8; ++j) {
    h0[j] = (_Float16)t0[j];
    h1[j] = (_Float16)t1[j];
  }
  _Float16* p = sb + n * SCR_STRIDE + hi * 8;
  *(v8h*)(p) = h0;        // K = hi*8 + j
  *(v8h*)(p + 16) = h1;   // K = 16 + hi*8 + j
  const _Float16* q = sb + n * SCR_STRIDE + hi * 16;
  v8h lo = *(const v8h*)(q);
  v8h hh = *(const v8h*)(q + 8);
  return __builtin_shufflevector(lo, hh, 0, 1, 2, 3, 4, 5, 6, 7, 8, 9, 10, 11,
                                 12, 13, 14, 15);
}

__device__ __forceinline__ void load16(float* d, const float* s) {
#pragma unroll
  for (int i = 0; i < 4; ++i) {
    float4 v = ((const float4*)s)[i];
    d[4 * i + 0] = v.x; d[4 * i + 1] = v.y;
    d[4 * i + 2] = v.z; d[4 * i + 3] = v.w;
  }
}

__device__ __forceinline__ void load48(float* d, const float* s) {
#pragma unroll
  for (int i = 0; i < 12; ++i) {
    float4 v = ((const float4*)s)[i];
    d[4 * i + 0] = v.x; d[4 * i + 1] = v.y;
    d[4 * i + 2] = v.z; d[4 * i + 3] = v.w;
  }
}

__device__ __forceinline__ v16h cvt16(const float* f) {
  v16h r;
#pragma unroll
  for (int i = 0; i < 16; ++i) r[i] = (_Float16)f[i];
  return r;
}

// Stage W (K x N, row-major f32) as WT (N x K, row-major f16) * scale in LDS.
__device__ __forceinline__ void stageT(_Float16* dst, const float* W, int K,
                                       int N, float scale, int tid, int nthr) {
  for (int i = tid; i < K * N; i += nthr) {
    int nn = i / K;
    int kk = i - nn * K;
    dst[i] = (_Float16)(W[kk * N + nn] * scale);
  }
}

__global__ __launch_bounds__(128, 1) void tp_kernel(
    const float* __restrict__ x1a, const float* __restrict__ x1b,
    const float* __restrict__ x2, const float* __restrict__ scal,
    const float* __restrict__ w0, const float* __restrict__ w1,
    const float* __restrict__ w2, const float* __restrict__ w3,
    const float* __restrict__ Wl0, const float* __restrict__ Wl1,
    const float* __restrict__ Wm1, const float* __restrict__ Wm2,
    const float* __restrict__ Wm3, const float* __restrict__ Wf0,
    const float* __restrict__ Wf1, float* __restrict__ out, int E) {
  __shared__ _Float16 lds[LDS_HALFS];
  {
    int tid = threadIdx.x, nthr = blockDim.x;
    stageT(lds + O_M1, Wm1, 64, 64, 0.125f, tid, nthr);
    stageT(lds + O_M2, Wm2, 64, 64, 0.125f * SILU_CST, tid, nthr);
    stageT(lds + O_M3, Wm3, 64, 128, 0.125f * SILU_CST * 0.125f, tid, nthr);
    stageT(lds + O_W0, w0, 64, 32, 1.f, tid, nthr);
    stageT(lds + O_W1, w1, 64, 32, 1.f, tid, nthr);
    stageT(lds + O_W2, w2, 64, 32, 1.f, tid, nthr);
    stageT(lds + O_W3, w3, 64, 32, 1.f, tid, nthr);
    stageT(lds + O_L0, Wl0, 64, 32, 0.125f, tid, nthr);
    stageT(lds + O_L1, Wl1, 64, 32, 0.125f, tid, nthr);
    stageT(lds + O_F0, Wf0, 32, 32, RSQ32, tid, nthr);
    stageT(lds + O_F1, Wf1, 32, 32, RSQ32, tid, nthr);
  }
  __syncthreads();

  const int lane = threadIdx.x & 31;
  const int wid = threadIdx.x >> 5;
  const int n = lane & 15;
  const int hi = lane >> 4;
  _Float16* scr0 = lds + O_SCR + wid * (2 * 640);
  _Float16* scr1 = scr0 + 640;
  const int ntiles = (E + 15) >> 4;
  const int nwaves = gridDim.x * 4;

  for (int t = blockIdx.x * 4 + wid; t < ntiles; t += nwaves) {
    const int e = t * 16 + n;                 // this lane's event (column)
    const long ee = (e < E) ? (long)e : (long)(E - 1);
    float fb[16];

    // ----- MLP gate: accW = PW * (Wm3' silu'(Wm2' silu'(Wm1'*scal'))) --------
    v16h Bsc0, Bsc1;
    {
      const float* sr = scal + ee * 64;
      load16(fb, sr + hi * 16);       Bsc0 = cvt16(fb);
      load16(fb, sr + 32 + hi * 16);  Bsc1 = cvt16(fb);
    }
    v8f accW[8];
    {
      v8f h[4];
#pragma unroll
      for (int mt = 0; mt < 4; ++mt) {
        v8f c = {};
        c = wmma_f16(lds_afrag(lds + O_M1, 64, mt, 0, n, hi), Bsc0, c);
        c = wmma_f16(lds_afrag(lds + O_M1, 64, mt, 1, n, hi), Bsc1, c);
#pragma unroll
        for (int j = 0; j < 8; ++j) {
          float v = c[j];                       // 1/sqrt(64) folded into Wm1
          c[j] = v / (1.f + __expf(-v));        // SILU_CST folded into Wm2
        }
        h[mt] = c;
      }
      v16h Bh0 = chunkToB(h[0], h[1], scr0, n, hi);
      v16h Bh1 = chunkToB(h[2], h[3], scr1, n, hi);
#pragma unroll
      for (int mt = 0; mt < 4; ++mt) {
        v8f c = {};
        c = wmma_f16(lds_afrag(lds + O_M2, 64, mt, 0, n, hi), Bh0, c);
        c = wmma_f16(lds_afrag(lds + O_M2, 64, mt, 1, n, hi), Bh1, c);
#pragma unroll
        for (int j = 0; j < 8; ++j) {
          float v = c[j];
          c[j] = v / (1.f + __expf(-v));        // SILU_CST folded into Wm3
        }
        h[mt] = c;
      }
      v16h Bg0 = chunkToB(h[0], h[1], scr0, n, hi);
      v16h Bg1 = chunkToB(h[2], h[3], scr1, n, hi);
#pragma unroll
      for (int mt = 0; mt < 8; ++mt) {          // PW and 1/8 folded into Wm3
        v8f c = {};
        c = wmma_f16(lds_afrag(lds + O_M3, 64, mt, 0, n, hi), Bg0, c);
        c = wmma_f16(lds_afrag(lds + O_M3, 64, mt, 1, n, hi), Bg1, c);
        accW[mt] = c;
      }
    }

    // ----- per-event geometric inputs (raw f16 fragments only) ---------------
    float y0v, y10, y11, y12;
    {
      const float4 xv = *(const float4*)(x2 + ee * 4);
      y0v = xv.x; y10 = xv.y; y11 = xv.z; y12 = xv.w;
    }
    const float ys0 = y10 * INV_S3, ys1 = y11 * INV_S3, ys2 = y12 * INV_S3;
    const float* ra = x1a + ee * 128;
    const float* rb = x1b + ee * 128;
    v16h Bs0_0, Bs0_1;
    {
      load16(fb, ra + hi * 16);  Bs0_0 = cvt16(fb);
      load16(fb, rb + hi * 16);  Bs0_1 = cvt16(fb);
    }
    v16h Bd1[2], Bs1[3][2];
    {
      float buf[48];
#pragma unroll
      for (int kc = 0; kc < 2; ++kc) {
        const float* r = (kc ? rb : ra) + 32 + 48 * hi;
        load48(buf, r);
        v16h d, q0, q1, q2;
#pragma unroll
        for (int i = 0; i < 16; ++i) {
          float a0 = buf[3 * i], a1 = buf[3 * i + 1], a2 = buf[3 * i + 2];
          d[i]  = (_Float16)(a0 * ys0 + a1 * ys1 + a2 * ys2);  // 1/sqrt3 folded
          q0[i] = (_Float16)a0;
          q1[i] = (_Float16)a1;
          q2[i] = (_Float16)a2;
        }
        Bd1[kc] = d;
        Bs1[0][kc] = q0; Bs1[1][kc] = q1; Bs1[2][kc] = q2;
      }
    }

    // ----- scalar (l=0) output path ------------------------------------------
    v8f m0t[2], m1t[2];
#pragma unroll
    for (int mt = 0; mt < 2; ++mt) {  // mid0 = y0 * (w0'*s0') * gate
      v8f c = {};
      c = wmma_f16(lds_afrag(lds + O_W0, 64, mt, 0, n, hi), Bs0_0, c);
      c = wmma_f16(lds_afrag(lds + O_W0, 64, mt, 1, n, hi), Bs0_1, c);
#pragma unroll
      for (int j = 0; j < 8; ++j) c[j] *= y0v * accW[mt][j];
      m0t[mt] = c;
    }
#pragma unroll
    for (int mt = 0; mt < 2; ++mt) {  // mid1 = (w1'*(s1.y1/sqrt3)') * gate
      v8f c = {};
      c = wmma_f16(lds_afrag(lds + O_W1, 64, mt, 0, n, hi), Bd1[0], c);
      c = wmma_f16(lds_afrag(lds + O_W1, 64, mt, 1, n, hi), Bd1[1], c);
#pragma unroll
      for (int j = 0; j < 8; ++j) c[j] *= accW[2 + mt][j];
      m1t[mt] = c;
    }
    v16h Bm0_0 = chunkToB(m0t[0], m0t[1], scr0, n, hi);
    v16h Bm0_1 = chunkToB(m1t[0], m1t[1], scr1, n, hi);
    v8f o0t[2];
#pragma unroll
    for (int mt = 0; mt < 2; ++mt) {  // o0 = Wl0'*m0' (1/8 folded into Wl0)
      v8f c = {};
      c = wmma_f16(lds_afrag(lds + O_L0, 64, mt, 0, n, hi), Bm0_0, c);
      c = wmma_f16(lds_afrag(lds + O_L0, 64, mt, 1, n, hi), Bm0_1, c);
      o0t[mt] = c;
    }
    v16h Bo0 = chunkToB(o0t[0], o0t[1], scr0, n, hi);
#pragma unroll
    for (int mt = 0; mt < 2; ++mt) {  // f0 = Wf0'*o0' (1/sqrt32 folded)
      v8f c = {};
      c = wmma_f16(lds_afrag(lds + O_F0, 32, mt, 0, n, hi), Bo0, c);
      if (e < E) {
        float* p = out + (long)e * 128 + mt * 16 + hi * 8;
        *(float4*)(p)     = make_float4(c[0], c[1], c[2], c[3]);
        *(float4*)(p + 4) = make_float4(c[4], c[5], c[6], c[7]);
      }
    }

    // ----- vector (l=1) output path ------------------------------------------
    v8f ptg[2], g3y[2];
#pragma unroll
    for (int mt = 0; mt < 2; ++mt) {  // p = w2'*s0', pre-gated once
      v8f c = {};
      c = wmma_f16(lds_afrag(lds + O_W2, 64, mt, 0, n, hi), Bs0_0, c);
      c = wmma_f16(lds_afrag(lds + O_W2, 64, mt, 1, n, hi), Bs0_1, c);
#pragma unroll
      for (int j = 0; j < 8; ++j) {
        c[j] *= accW[4 + mt][j];
        g3y[mt][j] = accW[6 + mt][j] * y0v;  // gate*y0 for mid3, once
      }
      ptg[mt] = c;
    }
#pragma unroll
    for (int cc = 0; cc < 3; ++cc) {
      const float y1c = (cc == 0) ? y10 : ((cc == 1) ? y11 : y12);
      v8f a[2], b[2];
#pragma unroll
      for (int mt = 0; mt < 2; ++mt) {  // mid2 = (p*gate) * y1c
        v8f c2;
#pragma unroll
        for (int j = 0; j < 8; ++j) c2[j] = ptg[mt][j] * y1c;
        a[mt] = c2;
      }
#pragma unroll
      for (int mt = 0; mt < 2; ++mt) {  // mid3 = (w3'*s1_c') * (gate*y0)
        v8f c = {};
        c = wmma_f16(lds_afrag(lds + O_W3, 64, mt, 0, n, hi), Bs1[cc][0], c);
        c = wmma_f16(lds_afrag(lds + O_W3, 64, mt, 1, n, hi), Bs1[cc][1], c);
#pragma unroll
        for (int j = 0; j < 8; ++j) c[j] *= g3y[mt][j];
        b[mt] = c;
      }
      v16h Bm1_0 = chunkToB(a[0], a[1], scr0, n, hi);
      v16h Bm1_1 = chunkToB(b[0], b[1], scr1, n, hi);
      v8f o1t[2];
#pragma unroll
      for (int mt = 0; mt < 2; ++mt) {  // o1_c = Wl1'*m1_c' (1/8 folded)
        v8f c = {};
        c = wmma_f16(lds_afrag(lds + O_L1, 64, mt, 0, n, hi), Bm1_0, c);
        c = wmma_f16(lds_afrag(lds + O_L1, 64, mt, 1, n, hi), Bm1_1, c);
        o1t[mt] = c;
      }
      v16h Bo1 = chunkToB(o1t[0], o1t[1], scr0, n, hi);
#pragma unroll
      for (int mt = 0; mt < 2; ++mt) {  // f1_c = Wf1'*o1_c' (1/sqrt32 folded)
        v8f c = {};
        c = wmma_f16(lds_afrag(lds + O_F1, 32, mt, 0, n, hi), Bo1, c);
        if (e < E) {
#pragma unroll
          for (int j = 0; j < 8; ++j)
            out[(long)e * 128 + 32 + 3 * (mt * 16 + hi * 8 + j) + cc] = c[j];
        }
      }
    }
  }
}

extern "C" void kernel_launch(void* const* d_in, const int* in_sizes, int n_in,
                              void* d_out, int out_size, void* d_ws,
                              size_t ws_size, hipStream_t stream) {
  const float* x1a = (const float*)d_in[0];
  const float* x1b = (const float*)d_in[1];
  const float* x2  = (const float*)d_in[2];
  const float* scl = (const float*)d_in[3];
  const float* w0  = (const float*)d_in[4];
  const float* w1  = (const float*)d_in[5];
  const float* w2  = (const float*)d_in[6];
  const float* w3  = (const float*)d_in[7];
  const float* Wl0 = (const float*)d_in[8];
  const float* Wl1 = (const float*)d_in[9];
  const float* Wm1 = (const float*)d_in[10];
  const float* Wm2 = (const float*)d_in[11];
  const float* Wm3 = (const float*)d_in[12];
  const float* Wf0 = (const float*)d_in[13];
  const float* Wf1 = (const float*)d_in[14];
  float* out = (float*)d_out;
  int E = in_sizes[0] / 128;

  dim3 grid(625), block(128);   // 2500 waves, one 16-event tile per iteration
  tp_kernel<<<grid, block, 0, stream>>>(x1a, x1b, x2, scl, w0, w1, w2, w3, Wl0,
                                        Wl1, Wm1, Wm2, Wm3, Wf0, Wf1, out, E);
}
